// LearnablePropositionExtractor_27668179321223
// MI455X (gfx1250) — matrix operand order repaired and checked by reading, not compiled, not run on hardware
//
#include <hip/hip_runtime.h>

typedef __attribute__((ext_vector_type(16))) _Float16 v16h;
typedef __attribute__((ext_vector_type(8)))  float    v8f;
typedef __attribute__((ext_vector_type(4)))  uint32_t v4u;

#define NUM_POS 17
#define FEAT    51          // 3 * 17
#define KPAD    64          // FEAT padded to 2 WMMA k-steps
#define HID1    128
#define HID2    64
#define SEQ     8192
#define SOUT    8190        // SEQ - 2
#define TILE_ROWS 64        // rows per A-tile (4 waves x 16)
#define TILES     4         // A-tiles per block (weights staged once)
#define ROWS_PER_BLOCK (TILE_ROWS * TILES)   // 256
#define THREADS 128
#define WAVES   4

union Frag16  { v16h h; uint32_t u[8]; v4u q[2]; };
union FragAcc { v8f  f; float    e[8]; };

static __device__ __forceinline__ uint32_t pack2f16(float a, float b) {
    _Float16 lo = (_Float16)a, hi = (_Float16)b;
    unsigned short ul, uh;
    __builtin_memcpy(&ul, &lo, 2);
    __builtin_memcpy(&uh, &hi, 2);
    return (uint32_t)ul | ((uint32_t)uh << 16);
}

__global__ __launch_bounds__(THREADS)
void prop_mlp_kernel(const float* __restrict__ pos, const float* __restrict__ W1,
                     const float* __restrict__ b1g, const float* __restrict__ W2,
                     const float* __restrict__ b2g, const float* __restrict__ W3,
                     const float* __restrict__ b3g, float* __restrict__ out)
{
    // ---- LDS (static, ~57 KB) ----
    __shared__ _Float16 sA[TILE_ROWS * KPAD];   // 8 KB  im2col tile, f16, K-padded
    // W1 fragment-major: [ks(2)][nt(8)][half(2)][l16(16)][v(8)] u32  (16 KB)
    __shared__ uint32_t sW1f[4096];
    // W2 fragment-major: [ks(4)][nt(4)][half(2)][l16(16)][v(8)] u32  (16 KB)
    __shared__ uint32_t sW2f[4096];
    __shared__ uint32_t sH[WAVES * 1024];       // 16 KB per-wave 4KB: h1(16x128 f16) / h2(16x64 f32)
    __shared__ float    sW3[HID2];
    __shared__ float    sB1[HID1];
    __shared__ float    sB2[HID2];
    __shared__ float    sB3;

    const int tid  = threadIdx.x;
    const int lane = tid & 31;
    const int wave = tid >> 5;
    const int half = (lane >> 4) & 1;   // 0: lanes 0-15, 1: lanes 16-31
    const int l16  = lane & 15;

    const int blocksPerB = (SOUT + ROWS_PER_BLOCK - 1) / ROWS_PER_BLOCK;   // 32
    const int b      = blockIdx.x / blocksPerB;
    const int sBase  = (blockIdx.x % blocksPerB) * ROWS_PER_BLOCK;

    // ---- stage W1 (51x128 -> 64x128 padded) in WMMA-B fragment order ----
    for (int i = tid; i < 4096; i += THREADS) {
        const int v  = i & 7;
        const int l  = (i >> 3) & 15;
        const int hf = (i >> 7) & 1;
        const int nt = (i >> 8) & 7;
        const int ks = (i >> 11) & 1;
        const int kp = ks * 16 + (v >> 2) * 8 + hf * 4 + (v & 3);       // f16 K-pair index
        const int n  = nt * 16 + l;
        const int k0 = 2 * kp, k1 = 2 * kp + 1;
        float lo = (k0 < FEAT) ? W1[k0 * HID1 + n] : 0.0f;
        float hi = (k1 < FEAT) ? W1[k1 * HID1 + n] : 0.0f;
        sW1f[i] = pack2f16(lo, hi);
    }
    // ---- stage W2 (128x64) in WMMA-B fragment order ----
    for (int i = tid; i < 4096; i += THREADS) {
        const int v  = i & 7;
        const int l  = (i >> 3) & 15;
        const int hf = (i >> 7) & 1;
        const int nt = (i >> 8) & 3;
        const int ks = (i >> 10) & 3;
        const int kp = ks * 16 + (v >> 2) * 8 + hf * 4 + (v & 3);
        const int n  = nt * 16 + l;
        sW2f[i] = pack2f16(W2[(2 * kp) * HID2 + n], W2[(2 * kp + 1) * HID2 + n]);
    }
    // ---- stage biases + W3 ----
    for (int i = tid; i < HID1; i += THREADS) sB1[i] = b1g[i];
    for (int i = tid; i < HID2; i += THREADS) { sB2[i] = b2g[i]; sW3[i] = W3[i]; }
    if (tid == 0) sB3 = b3g[0];
    __syncthreads();

    const int rowBase = wave * 16;                         // wave's 16-row sub-tile
    _Float16* h1 = (_Float16*)&sH[wave * 1024];            // 16 x 128 f16
    float*    h2 = (float*)&sH[wave * 1024];               // 16 x 64 f32 (overlay)

#pragma unroll 1
    for (int t = 0; t < TILES; ++t) {
        const int s0 = sBase + t * TILE_ROWS;

        // ---- stage im2col tile: 64 rows x 64 (padded) f16 ----
        for (int i = tid; i < TILE_ROWS * KPAD; i += THREADS) {
            const int r = i >> 6, k = i & 63;
            int s = s0 + r; if (s > SOUT - 1) s = SOUT - 1;   // clamp tail rows
            float v = 0.0f;
            if (k < FEAT) {
                const int w = k / NUM_POS;
                const int c = k - w * NUM_POS;
                v = pos[((size_t)b * SEQ + (size_t)(s + w)) * NUM_POS + c];
            }
            sA[r * KPAD + k] = (_Float16)v;
        }
        __syncthreads();

        // ================= Layer 1: [16 x 64] x [64 x 128] =================
        Frag16 a0, a1;
        {
            const v4u* arow = (const v4u*)&sA[(rowBase + l16) * KPAD];  // 8 quads / row
            a0.q[0] = arow[    half];
            a0.q[1] = arow[2 + half];
            a1.q[0] = arow[4 + half];
            a1.q[1] = arow[6 + half];
        }
#pragma unroll
        for (int nt = 0; nt < 8; ++nt) {
            const int n = nt * 16 + l16;
            const v4u* w0 = (const v4u*)&sW1f[(((0 * 8 + nt) * 2 + half) * 16 + l16) * 8];
            const v4u* w1 = (const v4u*)&sW1f[(((1 * 8 + nt) * 2 + half) * 16 + l16) * 8];
            Frag16 bf0, bf1;
            bf0.q[0] = w0[0]; bf0.q[1] = w0[1];
            bf1.q[0] = w1[0]; bf1.q[1] = w1[1];
            FragAcc c;
#pragma unroll
            for (int v = 0; v < 8; ++v) c.e[v] = 0.0f;
            c.f = __builtin_amdgcn_wmma_f32_16x16x32_f16(false, a0.h, false, bf0.h,
                                                         (short)0, c.f, false, false);
            c.f = __builtin_amdgcn_wmma_f32_16x16x32_f16(false, a1.h, false, bf1.h,
                                                         (short)0, c.f, false, false);
            const float bias = sB1[n];
#pragma unroll
            for (int v = 0; v < 8; ++v) {
                float x = fmaxf(c.e[v] + bias, 0.0f);          // ReLU
                h1[(half * 8 + v) * HID1 + n] = (_Float16)x;   // C layout -> row-major f16
            }
        }
        __syncthreads();

        // ================= Layer 2: [16 x 128] x [128 x 64] =================
        Frag16 a2[4];
        {
            const v4u* hr = (const v4u*)(h1 + l16 * HID1);     // 16 quads / row
#pragma unroll
            for (int ks = 0; ks < 4; ++ks) {
                a2[ks].q[0] = hr[ks * 4 +     half];
                a2[ks].q[1] = hr[ks * 4 + 2 + half];
            }
        }
        FragAcc c2[4];
#pragma unroll
        for (int nt = 0; nt < 4; ++nt) {
            const int n = nt * 16 + l16;
            FragAcc c;
#pragma unroll
            for (int v = 0; v < 8; ++v) c.e[v] = 0.0f;
#pragma unroll
            for (int ks = 0; ks < 4; ++ks) {
                const v4u* wf = (const v4u*)&sW2f[(((ks * 4 + nt) * 2 + half) * 16 + l16) * 8];
                Frag16 bf;
                bf.q[0] = wf[0]; bf.q[1] = wf[1];
                c.f = __builtin_amdgcn_wmma_f32_16x16x32_f16(false, a2[ks].h, false, bf.h,
                                                             (short)0, c.f, false, false);
            }
            const float bias = sB2[n];
#pragma unroll
            for (int v = 0; v < 8; ++v)
                c2[nt].e[v] = fmaxf(c.e[v] + bias, 0.0f);      // ReLU, keep in regs
        }
        __syncthreads();   // all h1 reads done before overlaying h2

#pragma unroll
        for (int nt = 0; nt < 4; ++nt)
#pragma unroll
            for (int v = 0; v < 8; ++v)
                h2[(half * 8 + v) * HID2 + (nt * 16 + l16)] = c2[nt].e[v];
        __syncthreads();

        // ================= Layer 3: [16 x 64] x [64 x 1] + sigmoid ==========
        if (lane < 16) {
            const float* hr = h2 + lane * HID2;
            float acc = sB3;
#pragma unroll 8
            for (int k = 0; k < HID2; ++k) acc = fmaf(hr[k], sW3[k], acc);
            const float y = 1.0f / (1.0f + __expf(-acc));
            const int s = s0 + rowBase + lane;
            if (s < SOUT) out[(size_t)b * SOUT + s] = y;
        }
        __syncthreads();   // h2 reads done before next tile reuses sA/sH
    }
}

extern "C" void kernel_launch(void* const* d_in, const int* in_sizes, int n_in,
                              void* d_out, int out_size, void* d_ws, size_t ws_size,
                              hipStream_t stream) {
    const float* pos = (const float*)d_in[0];
    const float* W1  = (const float*)d_in[1];
    const float* b1  = (const float*)d_in[2];
    const float* W2  = (const float*)d_in[3];
    const float* b2  = (const float*)d_in[4];
    const float* W3  = (const float*)d_in[5];
    const float* b3  = (const float*)d_in[6];
    float* out = (float*)d_out;

    const int blocksPerB = (SOUT + ROWS_PER_BLOCK - 1) / ROWS_PER_BLOCK;  // 32
    dim3 grid(64 * blocksPerB);
    dim3 block(THREADS);
    prop_mlp_kernel<<<grid, block, 0, stream>>>(pos, W1, b1, W2, b2, W3, b3, out);
}
